// MoE_11098195493463
// MI455X (gfx1250) — matrix operand order repaired and checked by reading, not compiled, not run on hardware
//
#include <hip/hip_runtime.h>
#include <hip/hip_bf16.h>
#include <math.h>

#define T_ 4096
#define C_ 1024
#define H_ 4096
#define E_ 8
#define K_ 2

typedef _Float16 h16;
typedef __attribute__((ext_vector_type(16))) _Float16 v16h;
typedef __attribute__((ext_vector_type(8)))  _Float16 v8h;
typedef __attribute__((ext_vector_type(8)))  float    v8f;
typedef __attribute__((ext_vector_type(4)))  int      v4i_;

// LDS strides (in halves), padded; every row offset is 16B-aligned.
#define XS_LD  1032   // xs: 16 x 1024 (+8 pad)
#define W1T_LD 136    // w1t: [h_local][k], 128 x 128 (+8 pad)
#define W2T_LD 40     // w2t: [c][k], 1024 x 32 (+8 pad)
#define HB_LD  136    // hb: 16 x 128 (+8 pad)
#define SMEM_HALVES (16 * XS_LD + 128 * W1T_LD + 1024 * W2T_LD + 16 * HB_LD)
#define SMEM_BYTES  (SMEM_HALVES * 2)

// ---------------- CDNA5 async global->LDS copy (ASYNCcnt path) ----------------
#if __has_builtin(__builtin_amdgcn_global_load_async_to_lds_b128)
#define HAVE_ASYNC_LDS 1
#else
#define HAVE_ASYNC_LDS 0
#endif

__device__ __forceinline__ void cp_b128(const h16* g, h16* l) {
#if HAVE_ASYNC_LDS
  __builtin_amdgcn_global_load_async_to_lds_b128(
      (__attribute__((address_space(1))) v4i_*)(unsigned long long)g,
      (__attribute__((address_space(3))) v4i_*)(unsigned int)(unsigned long long)l,
      0, 0);
#else
  *(v8h*)l = *(const v8h*)g;
#endif
}
__device__ __forceinline__ void cp_wait() {
#if HAVE_ASYNC_LDS
#if __has_builtin(__builtin_amdgcn_s_wait_asynccnt)
  __builtin_amdgcn_s_wait_asynccnt(0);
#else
  asm volatile("s_wait_asynccnt 0x0" ::: "memory");
#endif
#endif
}

// ---------- WMMA fragment loaders (ISA 16-bit layouts, wave32) ----------
// A 16x32 f16: lanes 0-15: M=lane, K {0..7,16..23}; lanes 16-31: M=lane-16, K {8..15,24..31}
__device__ __forceinline__ v16h load_a16(const h16* base, int ld, int kc, int lane) {
  int m  = lane & 15;
  int kb = (lane & 16) ? 8 : 0;
  const h16* p = base + m * ld + kc + kb;
  v8h lo = *(const v8h*)(p);
  v8h hi = *(const v8h*)(p + 16);
  return __builtin_shufflevector(lo, hi, 0,1,2,3,4,5,6,7,8,9,10,11,12,13,14,15);
}
// B 32x16 f16 stored transposed in LDS: bt[n*ld + k].
// lanes 0-15: N=lane, K {0..7,16..23}; lanes 16-31: N=lane-16, K {8..15,24..31}
__device__ __forceinline__ v16h load_bt16(const h16* bt, int ld, int n0, int kc, int lane) {
  int n  = lane & 15;
  int k0 = (lane & 16) ? 8 : 0;
  const h16* p = bt + (n0 + n) * ld + kc + k0;
  v8h lo = *(const v8h*)(p);
  v8h hi = *(const v8h*)(p + 16);
  return __builtin_shufflevector(lo, hi, 0,1,2,3,4,5,6,7,8,9,10,11,12,13,14,15);
}

// ---------- kernel 0: fp32 -> f16 weight/activation pre-pass ----------
// W1 (E,C,H) -> W1T (E,H,C) f16 ; W2 (E,H,C) -> W2T (E,C,H) f16 ; x -> f16
#define NW_ ((long)E_ * C_ * H_ / 4)   // quads per weight tensor
#define NX_ ((long)T_ * C_ / 4)        // quads for x
__global__ __launch_bounds__(256) void moe_conv_kernel(
    const float* __restrict__ W1, const float* __restrict__ W2,
    const float* __restrict__ x,
    h16* __restrict__ W1T, h16* __restrict__ W2T, h16* __restrict__ xh) {
  long j = (long)blockIdx.x * 256 + threadIdx.x;
  if (j < NW_) {                       // W1T quad along c (dest-major)
    long pos = j * 4;
    int e = (int)(pos >> 22);          // C_*H_ == 2^22
    long rem = pos & ((1L << 22) - 1);
    int h = (int)(rem >> 10), c = (int)(rem & (C_ - 1));
    h16* d = W1T + pos;
#pragma unroll
    for (int i = 0; i < 4; ++i)
      d[i] = (h16)W1[((long)e * C_ + c + i) * H_ + h];
  } else if (j < 2 * NW_) {            // W2T quad along h (dest-major)
    long pos = (j - NW_) * 4;
    int e = (int)(pos >> 22);
    long rem = pos & ((1L << 22) - 1);
    int c = (int)(rem >> 12), h = (int)(rem & (H_ - 1));
    h16* d = W2T + pos;
#pragma unroll
    for (int i = 0; i < 4; ++i)
      d[i] = (h16)W2[((long)e * H_ + h + i) * C_ + c];
  } else if (j < 2 * NW_ + NX_) {      // x straight conversion (coalesced)
    long pos = (j - 2 * NW_) * 4;
    const float4 v = *(const float4*)(x + pos);
    h16* d = xh + pos;
    d[0] = (h16)v.x; d[1] = (h16)v.y; d[2] = (h16)v.z; d[3] = (h16)v.w;
  }
}

// ---------- kernel 1: zero comb (d_out) + counters ----------
__global__ __launch_bounds__(256) void moe_zero_kernel(float* __restrict__ comb,
                                                       int* __restrict__ cnt,
                                                       float* __restrict__ imp,
                                                       float* __restrict__ loadv) {
  int idx = blockIdx.x * 256 + threadIdx.x;
  for (long i = idx; i < (long)T_ * C_; i += (long)gridDim.x * 256) comb[i] = 0.f;
  if (idx < E_) { cnt[idx] = 0; imp[idx] = 0.f; loadv[idx] = 0.f; }
}

// ---------- kernel 2: gating (one wave32 per token) ----------
__global__ __launch_bounds__(256) void moe_gate_kernel(
    const float* __restrict__ x, const float* __restrict__ ctx,
    const float* __restrict__ noise, const float* __restrict__ w_gate,
    const float* __restrict__ w_noise,
    int* __restrict__ pt_idx, float* __restrict__ pt_g,
    float* __restrict__ imp, float* __restrict__ loadv) {
  int tid  = threadIdx.x;
  int lane = tid & 31;
  int wv   = tid >> 5;
  int t    = blockIdx.x * 8 + wv;
  int e = lane & 7;       // expert handled by this lane
  int j = lane >> 3;      // K-partition 0..3

  float cg = 0.f, cn = 0.f;
  for (int c = j; c < 2 * C_; c += 4) {
    float gv = (c < C_) ? x[(long)t * C_ + c] : ctx[(long)t * C_ + (c - C_)];
    cg += gv * w_gate[(long)c * E_ + e];
    cn += gv * w_noise[(long)c * E_ + e];
  }
  cg += __shfl_xor(cg, 8, 32);  cg += __shfl_xor(cg, 16, 32);
  cn += __shfl_xor(cn, 8, 32);  cn += __shfl_xor(cn, 16, 32);

  float sp    = (cn > 20.f) ? cn : log1pf(expf(cn));
  float sigma = sp + 0.01f;                       // NOISE_EPS
  float logit = cg + noise[(long)t * E_ + e] * sigma;

  float v[8];
#pragma unroll
  for (int q = 0; q < 8; ++q) v[q] = __shfl(logit, q, 32);

  int i0 = 0; float v0 = v[0];
#pragma unroll
  for (int q = 1; q < 8; ++q) if (v[q] > v0) { v0 = v[q]; i0 = q; }
  int i1 = -1; float v1 = -1e30f;
#pragma unroll
  for (int q = 0; q < 8; ++q) if (q != i0 && v[q] > v1) { v1 = v[q]; i1 = q; }
  float v2 = -1e30f;
#pragma unroll
  for (int q = 0; q < 8; ++q) if (q != i0 && q != i1 && v[q] > v2) v2 = v[q];

  float g0 = 1.f / (1.f + expf(v1 - v0));   // softmax over {v0, v1}
  float g1 = 1.f - g0;

  if (lane < 8) {
    bool  is_in = logit > v2;                 // thr_in = 3rd largest
    float thr   = is_in ? v2 : v1;            // else thr_out = 2nd largest
    float z     = (cg - thr) / sigma;
    float prob  = 0.5f * erfcf(-z * 0.70710678118654752f);
    atomicAdd(&loadv[lane], prob);
  }
  if (lane == 0) {
    atomicAdd(&imp[i0], g0);
    atomicAdd(&imp[i1], g1);
    pt_idx[t * K_ + 0] = i0;  pt_g[t * K_ + 0] = g0;
    pt_idx[t * K_ + 1] = i1;  pt_g[t * K_ + 1] = g1;
  }
}

// ---------- kernel 3: scatter pairs into per-expert buckets ----------
__global__ __launch_bounds__(256) void moe_scatter_kernel(
    const int* __restrict__ pt_idx, const float* __restrict__ pt_g,
    int* __restrict__ cnt, int* __restrict__ bucket, float* __restrict__ bgate) {
  int i = blockIdx.x * 256 + threadIdx.x;
  if (i < T_ * K_) {
    int e = pt_idx[i];
    int t = i >> 1;
    int pos = atomicAdd(&cnt[e], 1);
    bucket[e * T_ + pos] = t;
    bgate[e * T_ + pos]  = pt_g[i];
  }
}

// ---------- kernel 4: aux loss ----------
__global__ void moe_loss_kernel(const float* __restrict__ imp,
                                const float* __restrict__ loadv,
                                float* __restrict__ out_loss) {
  if (threadIdx.x == 0 && blockIdx.x == 0) {
    float mi = 0.f, ml = 0.f;
    for (int e = 0; e < E_; ++e) { mi += imp[e]; ml += loadv[e]; }
    mi *= (1.f / E_); ml *= (1.f / E_);
    float vi = 0.f, vl = 0.f;
    for (int e = 0; e < E_; ++e) {
      float d = imp[e] - mi;  vi += d * d;
      float f = loadv[e] - ml; vl += f * f;
    }
    vi *= (1.f / (E_ - 1)); vl *= (1.f / (E_ - 1));
    float cvi = vi / (mi * mi + 1e-10f);
    float cvl = vl / (ml * ml + 1e-10f);
    out_loss[0] = (cvi + cvl) * 0.01f;
  }
}

// ---------- kernel 5: routed expert FFN with WMMA + async LDS staging ----------
// grid: E_*256 blocks; block = (expert e, row tile rt of 16 routed tokens)
// 8 wave32s; each wave owns 8 output 16x16 tiles (128 of the 1024 columns).
__global__ __launch_bounds__(256) void moe_ffn_kernel(
    const h16* __restrict__ xh,
    const h16* __restrict__ W1T, const float* __restrict__ b1,
    const h16* __restrict__ W2T, const float* __restrict__ b2,
    const int* __restrict__ cnt, const int* __restrict__ bucket,
    const float* __restrict__ bgate, float* __restrict__ comb) {
  extern __shared__ char smem[];
  h16* xs  = (h16*)smem;                 // [16][XS_LD]
  h16* w1t = xs + 16 * XS_LD;            // [128][W1T_LD]  W1 block as [h][k]
  h16* w2t = w1t + 128 * W1T_LD;         // [1024][W2T_LD] W2 block as [c][k]
  h16* hb  = w2t + 1024 * W2T_LD;        // [16][HB_LD]    gelu(h) tile
  __shared__ int   toks[16];
  __shared__ float gts[16];

  int e  = blockIdx.x >> 8;
  int rt = blockIdx.x & 255;
  int n  = cnt[e];
  int r0 = rt * 16;
  if (r0 >= n) return;
  int valid = n - r0; if (valid > 16) valid = 16;

  int tid  = threadIdx.x;
  int lane = tid & 31;
  int wv   = tid >> 5;
  int nloc = lane & 15;

  if (tid < 16) {
    int r = tid;
    toks[r] = (r < valid) ? bucket[e * T_ + r0 + r] : 0;
    gts[r]  = (r < valid) ? bgate[e * T_ + r0 + r] : 0.f;
  }
  __syncthreads();

  // stage token activations (f16, async b128: 8 chunks/thread)
  for (int q = tid; q < 16 * 128; q += 256) {
    int row = q >> 7, ch = q & 127;
    cp_b128(xh + (long)toks[row] * C_ + ch * 8, xs + row * XS_LD + ch * 8);
  }
  cp_wait();

  const h16*  W1Te = W1T + (long)e * H_ * C_;   // [h][c]
  const h16*  W2Te = W2T + (long)e * C_ * H_;   // [c][h]
  const float* b1e = b1 + e * H_;
  const float* b2e = b2 + e * C_;

  v8f acc[8];
#pragma unroll
  for (int jj = 0; jj < 8; ++jj) { v8f z = {}; acc[jj] = z; }

  for (int hh = 0; hh < H_; hh += 128) {
    // ---- GEMM1: ps(16x16) = xs(16x1024) @ W1[:, hh + wv*16 .. +16)
    v8f ps = {};
    for (int kc = 0; kc < C_; kc += 128) {
      __syncthreads();   // previous consumers of w1t done
      // async-stage W1T rows hh..hh+127, k-range kc..kc+127 (8 chunks/thread)
      for (int q = tid; q < 128 * 16; q += 256) {
        int row = q >> 4, ch = q & 15;
        cp_b128(W1Te + (long)(hh + row) * C_ + kc + ch * 8,
                w1t + row * W1T_LD + ch * 8);
      }
      // prefetch next k-stage (decorative hint -> global_prefetch_b8)
      __builtin_prefetch(W1Te + (long)(hh + (tid >> 1)) * C_ +
                         ((kc + 128) & (C_ - 1)) + (tid & 1) * 64, 0, 0);
      cp_wait();
      __syncthreads();
      v16h ga[4], gb[4];
#pragma unroll
      for (int kk = 0; kk < 4; ++kk) {
        ga[kk] = load_a16(xs, XS_LD, kc + kk * 32, lane);
        gb[kk] = load_bt16(w1t, W1T_LD, wv * 16, kk * 32, lane);
      }
#pragma unroll
      for (int kk = 0; kk < 4; ++kk)
        ps = __builtin_amdgcn_wmma_f32_16x16x32_f16(false, ga[kk], false, gb[kk],
                                                    (short)0, ps, false, false);
    }
    // bias + exact GELU -> hb (f16). D layout: VGPR r -> M = r + 8*(lane>=16)
    float bb = b1e[hh + wv * 16 + nloc];
    __syncthreads();   // previous GEMM2 reads of hb done
#pragma unroll
    for (int r = 0; r < 8; ++r) {
      int m = r + ((lane >> 4) << 3);
      float v = ps[r] + bb;
      float g = 0.5f * v * (1.f + erff(v * 0.70710678118654752f));
      hb[m * HB_LD + wv * 16 + nloc] = (h16)g;
    }
    // ---- GEMM2: acc += hb(16x128) @ W2[hh..hh+128, :], 4 k-substages of 32
    for (int s = 0; s < 4; ++s) {
      __syncthreads();  // hb visible (s==0) / previous w2t consumers done
      // async-stage W2T rows c=0..1023, k-range (hh+s*32 .. +32) (16 chunks/thread)
      for (int q = tid; q < 1024 * 4; q += 256) {
        int row = q >> 2, ch = q & 3;
        cp_b128(W2Te + (long)row * H_ + hh + s * 32 + ch * 8,
                w2t + row * W2T_LD + ch * 8);
      }
      __builtin_prefetch(W2Te + (long)(tid * 4) * H_ + hh + ((s + 1) & 3) * 32, 0, 0);
      cp_wait();
      __syncthreads();
      v16h a = load_a16(hb, HB_LD, s * 32, lane);
      v16h bfr[8];
#pragma unroll
      for (int jj = 0; jj < 8; ++jj)
        bfr[jj] = load_bt16(w2t, W2T_LD, wv * 128 + jj * 16, 0, lane);
#pragma unroll
      for (int jj = 0; jj < 8; ++jj)
        acc[jj] = __builtin_amdgcn_wmma_f32_16x16x32_f16(false, a, false, bfr[jj],
                                                         (short)0, acc[jj], false, false);
    }
  }

  // ---- epilogue: comb[t, col] += gate * exp(out + b2)
#pragma unroll
  for (int r = 0; r < 8; ++r) {
    int m = r + ((lane >> 4) << 3);
    if (m < valid) {
      int   t = toks[m];
      float g = gts[m];
#pragma unroll
      for (int jj = 0; jj < 8; ++jj) {
        int col = wv * 128 + jj * 16 + nloc;
        float v = acc[jj][r] + b2e[col];
        atomicAdd(comb + (long)t * C_ + col, g * expf(v));
      }
    }
  }
}

// ---------- kernel 6: y = log(comb) in place ----------
__global__ __launch_bounds__(256) void moe_log_kernel(float* __restrict__ comb) {
  long i = (long)blockIdx.x * 256 + threadIdx.x;
  if (i < (long)T_ * C_) {
    float v = comb[i];
    comb[i] = logf(v == 0.f ? 2.2204460492503131e-16f : v);
  }
}

extern "C" void kernel_launch(void* const* d_in, const int* in_sizes, int n_in,
                              void* d_out, int out_size, void* d_ws, size_t ws_size,
                              hipStream_t stream) {
  const float* x       = (const float*)d_in[0];
  const float* ctx     = (const float*)d_in[1];
  const float* noise   = (const float*)d_in[2];
  const float* w_gate  = (const float*)d_in[3];
  const float* w_noise = (const float*)d_in[4];
  const float* W1      = (const float*)d_in[5];
  const float* b1      = (const float*)d_in[6];
  const float* W2      = (const float*)d_in[7];
  const float* b2      = (const float*)d_in[8];
  float* out = (float*)d_out;   // [T*C] y (used as comb accumulator) + [1] loss

  // ws layout: small control block, then f16 tensors at 512KB offset
  float* wsf        = (float*)d_ws;
  int*   cnt        = (int*)wsf;                      // E
  float* importance = wsf + 8;                        // E
  float* loadv      = wsf + 16;                       // E
  int*   pt_idx     = (int*)(wsf + 24);               // T*K
  float* pt_g       = wsf + 24 + T_ * K_;             // T*K
  int*   bucket     = (int*)(wsf + 24 + 2 * T_ * K_); // E*T
  float* bgate      = wsf + 24 + 2 * T_ * K_ + E_ * T_;
  h16*   W1T        = (h16*)((char*)d_ws + (1 << 19));        // E*H*C halves
  h16*   W2T        = W1T + (long)E_ * H_ * C_;               // E*C*H halves
  h16*   xh         = W2T + (long)E_ * C_ * H_;               // T*C halves

  long conv_threads = 2 * NW_ + NX_;
  int  conv_grid    = (int)((conv_threads + 255) / 256);

  moe_conv_kernel<<<conv_grid, 256, 0, stream>>>(W1, W2, x, W1T, W2T, xh);
  moe_zero_kernel<<<1024, 256, 0, stream>>>(out, cnt, importance, loadv);
  moe_gate_kernel<<<T_ / 8, 256, 0, stream>>>(x, ctx, noise, w_gate, w_noise,
                                              pt_idx, pt_g, importance, loadv);
  moe_scatter_kernel<<<(T_ * K_) / 256, 256, 0, stream>>>(pt_idx, pt_g, cnt, bucket, bgate);
  moe_loss_kernel<<<1, 32, 0, stream>>>(importance, loadv, out + (long)T_ * C_);
  moe_ffn_kernel<<<E_ * 256, 256, SMEM_BYTES, stream>>>(xh, W1T, b1, W2T, b2,
                                                        cnt, bucket, bgate, out);
  moe_log_kernel<<<(T_ * C_) / 256, 256, 0, stream>>>(out);
}
